// contrast_loss_18726057411399
// MI455X (gfx1250) — matrix operand order repaired and checked by reading, not compiled, not run on hardware
//
#include <hip/hip_runtime.h>
#include <hip/hip_bf16.h>

// ---------------------------------------------------------------------------
// Problem constants (reference: N=8192 samples, D=512, 2 views)
// ---------------------------------------------------------------------------
#define NROWS 16384        // 2*N rows of z
#define DIM   512          // embedding dim
#define NPAIR 4097         // n/2 + 1 pairs indexed by the reference
#define BM    128          // output tile rows per workgroup
#define BN    128          // output tile cols per workgroup
#define KB    32           // K per WMMA (f16 16x16x32)
#define LDP   40           // LDS row pitch in halves (80B: 16B-aligned, conflict-free)

typedef _Float16 v16h __attribute__((ext_vector_type(16)));
typedef _Float16 v8h  __attribute__((ext_vector_type(8)));
typedef float    v8f  __attribute__((ext_vector_type(8)));
typedef int      v4i  __attribute__((vector_size(16)));   // builtin's pointee type

#define AS1 __attribute__((address_space(1)))
#define AS3 __attribute__((address_space(3)))

#if __has_builtin(__builtin_amdgcn_global_load_async_to_lds_b128)
#define HAS_ASYNC_LDS 1
#else
#define HAS_ASYNC_LDS 0
#endif

__device__ __forceinline__ float waveReduceSum(float v) {
#pragma unroll
    for (int m = 16; m >= 1; m >>= 1) v += __shfl_xor(v, m, 32);
    return v;
}

// One 16B chunk global -> LDS. Async (ASYNCcnt-tracked DMA path) when the
// toolchain exposes the gfx1250 builtin; synchronous load + ds_store fallback.
__device__ __forceinline__ void copy16_to_lds(void* l, const void* g) {
#if HAS_ASYNC_LDS
    // AS3 pointer value == 32-bit wave-relative LDS address (flat aperture rule)
    AS1 v4i* gp = (AS1 v4i*)(unsigned long long)(uintptr_t)g;
    AS3 v4i* lp = (AS3 v4i*)(unsigned)(uintptr_t)l;
    __builtin_amdgcn_global_load_async_to_lds_b128(gp, lp, 0, 0);
#else
    *(v8h*)l = *(const v8h*)g;
#endif
}

__device__ __forceinline__ void wait_async_le4() {
#if HAS_ASYNC_LDS
#if __has_builtin(__builtin_amdgcn_s_wait_asynccnt)
    __builtin_amdgcn_s_wait_asynccnt(4);
#else
    asm volatile("s_wait_asynccnt 0x4" ::: "memory");
#endif
#endif
}
__device__ __forceinline__ void wait_async_0() {
#if HAS_ASYNC_LDS
#if __has_builtin(__builtin_amdgcn_s_wait_asynccnt)
    __builtin_amdgcn_s_wait_asynccnt(0);
#else
    asm volatile("s_wait_asynccnt 0x0" ::: "memory");
#endif
#endif
}

// ---------------------------------------------------------------------------
// Kernel 1: de-interleave + L2-normalize rows of z, store FP16 zn (16 MB,
// fully L2-resident: all GEMM traffic after this pass stays on-chip).
// Also emit per-row diag term exp(||zn16_i||^2 / T) consistent with the
// FP16 Gram matrix, for the trace subtraction.
// ---------------------------------------------------------------------------
__global__ __launch_bounds__(256)
void normalize_rows(const float* __restrict__ y, const float* __restrict__ temp,
                    _Float16* __restrict__ zn, float* __restrict__ diag) {
    const int wave = threadIdx.x >> 5;
    const int lane = threadIdx.x & 31;
    const int row  = blockIdx.x * 8 + wave;
    if (row >= NROWS) return;

    // z[row][d] = y[row>>1][d][row&1]  (y is (N, D, 2) row-major)
    const float* src = y + (size_t)(row >> 1) * (2 * DIM) + (row & 1);

    float vals[16];
    float ss = 0.f;
#pragma unroll
    for (int t = 0; t < 16; ++t) {
        const int d = t * 32 + lane;
        vals[t] = src[(size_t)d * 2];
        ss += vals[t] * vals[t];
    }
    ss = waveReduceSum(ss);
    float nrm = __builtin_amdgcn_sqrtf(ss);
    nrm = fmaxf(nrm, 1e-8f);                     // torch CosineSimilarity eps clamp
    const float inv = 1.0f / nrm;

    _Float16* dst = zn + (size_t)row * DIM;
    float ss16 = 0.f;
#pragma unroll
    for (int t = 0; t < 16; ++t) {
        const int d = t * 32 + lane;
        const _Float16 h = (_Float16)(vals[t] * inv);
        dst[d] = h;
        const float f = (float)h;
        ss16 += f * f;
    }
    ss16 = waveReduceSum(ss16);
    if (lane == 0) {
        const float scale = 1.4426950408889634f / temp[0];   // log2(e)/T
        diag[row] = __builtin_amdgcn_exp2f(ss16 * scale);
    }
}

// ---------------------------------------------------------------------------
// Kernel 2: Gram(zn) tile -> exp -> block-local sum, upper triangle only
// (exact by symmetry: sim[i,j] and sim[j,i] use identical MAC ordering).
// 128x128 tile per 256-thread block; 8 waves (2x4), each wave 4x2 WMMA accums.
// A/B 128x32 tiles double-buffered in LDS (rows padded to 40 halves so the
// 16-lane ds_load_b128 fragment reads hit 16 distinct banks). Tiles are
// filled with GLOBAL_LOAD_ASYNC_TO_LDS_B128 (ASYNCcnt) so tile k+1 streams
// in behind the v_wmma work on tile k.
// Fragment layouts per CDNA5 ISA:
//   A (16x32 f16): lane l -> row m=l&15; halves[0..7]=K (l>=16?8:0)+0..7,
//                  halves[8..15]=K 16+(l>=16?8:0)+0..7
//   B (32x16 f16): lane l -> col n=l&15; halves[0..7]=K (l>=16?8:0)+0..7,
//                  halves[8..15]=K 16+(l>=16?8:0)+0..7
// ---------------------------------------------------------------------------
__global__ __launch_bounds__(256)
void gram_exp_sum(const _Float16* __restrict__ zn, const float* __restrict__ temp,
                  float* __restrict__ partials) {
    const int iTile = blockIdx.y;
    const int jTile = blockIdx.x;
    const int pidx  = iTile * gridDim.x + jTile;
    if (jTile < iTile) {                         // lower triangle: contribute 0
        if (threadIdx.x == 0) partials[pidx] = 0.f;
        return;
    }

    __shared__ alignas(16) _Float16 As[2][BM][LDP];
    __shared__ alignas(16) _Float16 Bs[2][BN][LDP];

    const int tid  = threadIdx.x;
    const int wave = tid >> 5;
    const int lane = tid & 31;
    const int wr   = wave >> 2;                  // 0..1 (64 rows)
    const int wc   = wave & 3;                   // 0..3 (32 cols)
    const int l15  = lane & 15;
    const int hsel = lane >> 4;
    const int iBase = iTile * BM;
    const int jBase = jTile * BN;

    // Cooperative tile fill: 512 chunks of 16B per matrix, 2 per thread each.
    auto load_tiles = [&](int ks, int buf) {
        const int k0 = ks * KB;
#pragma unroll
        for (int q = 0; q < 2; ++q) {
            const int c  = tid + q * 256;
            const int r  = c >> 2;               // 0..127
            const int cc = (c & 3) * 8;          // half offset 0/8/16/24
            copy16_to_lds(&As[buf][r][cc], zn + (size_t)(iBase + r) * DIM + k0 + cc);
            copy16_to_lds(&Bs[buf][r][cc], zn + (size_t)(jBase + r) * DIM + k0 + cc);
        }
    };

    v8f c[4][2] = {};
    const int NK = DIM / KB;                     // 16 K-steps

    load_tiles(0, 0);                            // prologue
    for (int ks = 0; ks < NK; ++ks) {
        const int cur = ks & 1;
        if (ks + 1 < NK) {
            load_tiles(ks + 1, cur ^ 1);         // stream next tile behind compute
            wait_async_le4();                    // older 4 (cur tile) complete
        } else {
            wait_async_0();
        }
        __syncthreads();                         // cur tiles visible WGP-wide

        v16h a[4], b[2];
#pragma unroll
        for (int mt = 0; mt < 4; ++mt) {
            const int r = wr * 64 + mt * 16 + l15;
            const v8h lo = *(const v8h*)&As[cur][r][hsel * 8];
            const v8h hi = *(const v8h*)&As[cur][r][16 + hsel * 8];
#pragma unroll
            for (int i = 0; i < 8; ++i) { a[mt][i] = lo[i]; a[mt][i + 8] = hi[i]; }
        }
#pragma unroll
        for (int nt = 0; nt < 2; ++nt) {
            const int r = wc * 32 + nt * 16 + l15;
            const v8h lo = *(const v8h*)&Bs[cur][r][hsel * 8];
            const v8h hi = *(const v8h*)&Bs[cur][r][16 + hsel * 8];
#pragma unroll
            for (int i = 0; i < 8; ++i) { b[nt][i] = lo[i]; b[nt][i + 8] = hi[i]; }
        }
#pragma unroll
        for (int mt = 0; mt < 4; ++mt)
#pragma unroll
            for (int nt = 0; nt < 2; ++nt)
                c[mt][nt] = __builtin_amdgcn_wmma_f32_16x16x32_f16(
                    false, a[mt], false, b[nt], (short)0, c[mt][nt], false, false);

        __syncthreads();                         // reads done before buf reuse
    }

    // Fused exp(sim/T) + reduction (v_exp_f32 on the VALU pipe)
    const float scale = 1.4426950408889634f / temp[0];    // log2(e)/T
    float sum = 0.f;
#pragma unroll
    for (int mt = 0; mt < 4; ++mt)
#pragma unroll
        for (int nt = 0; nt < 2; ++nt)
#pragma unroll
            for (int e = 0; e < 8; ++e)
                sum += __builtin_amdgcn_exp2f(c[mt][nt][e] * scale);

    sum = waveReduceSum(sum);
    __shared__ float red[8];
    if (lane == 0) red[wave] = sum;
    __syncthreads();
    if (tid == 0) {
        float s = 0.f;
#pragma unroll
        for (int w = 0; w < 8; ++w) s += red[w];
        const float weight = (iTile == jTile) ? 1.0f : 2.0f;  // symmetry fold
        partials[pidx] = s * weight;
    }
}

// ---------------------------------------------------------------------------
// Kernel 3: the 4097 reference pair dots sim[2p, 2p+1]
// ---------------------------------------------------------------------------
__global__ __launch_bounds__(256)
void pair_dots(const _Float16* __restrict__ zn, float* __restrict__ pd) {
    const int wave = threadIdx.x >> 5;
    const int lane = threadIdx.x & 31;
    const int p = blockIdx.x * 8 + wave;
    if (p >= NPAIR) return;
    const _Float16* r0 = zn + (size_t)(2 * p) * DIM;
    const _Float16* r1 = r0 + DIM;
    float s = 0.f;
#pragma unroll
    for (int t = 0; t < 16; ++t) {
        const int d = t * 32 + lane;
        s += (float)r0[d] * (float)r1[d];
    }
    s = waveReduceSum(s);
    if (lane == 0) pd[p] = s;
}

// ---------------------------------------------------------------------------
// Kernel 4: deterministic final reduction + closed-form loss.
// result = ( P*log(denom) - (1/T)*sum(pair_dots) ) / (ln10 * n)
// denom  = sum(exp(sim/T)) - trace(exp(sim/T))
// ---------------------------------------------------------------------------
__global__ __launch_bounds__(256)
void finalize(const float* __restrict__ gp, int ngp,
              const float* __restrict__ diag,
              const float* __restrict__ pd,
              const float* __restrict__ temp,
              float* __restrict__ out) {
    __shared__ float red[256];
    const int tid = threadIdx.x;

    float s = 0.f;
    for (int i = tid; i < ngp; i += 256) s += gp[i];
    red[tid] = s; __syncthreads();
    for (int off = 128; off > 0; off >>= 1) { if (tid < off) red[tid] += red[tid + off]; __syncthreads(); }
    const float Sall = red[0]; __syncthreads();

    s = 0.f;
    for (int i = tid; i < NROWS; i += 256) s += diag[i];
    red[tid] = s; __syncthreads();
    for (int off = 128; off > 0; off >>= 1) { if (tid < off) red[tid] += red[tid + off]; __syncthreads(); }
    const float Sdiag = red[0]; __syncthreads();

    s = 0.f;
    for (int i = tid; i < NPAIR; i += 256) s += pd[i];
    red[tid] = s; __syncthreads();
    for (int off = 128; off > 0; off >>= 1) { if (tid < off) red[tid] += red[tid + off]; __syncthreads(); }
    const float Spair = red[0];

    if (tid == 0) {
        const float T     = temp[0];
        const float denom = Sall - Sdiag;
        const float logd  = __builtin_amdgcn_logf(denom) * 0.6931471805599453f; // ln
        const float LN10  = 2.302585092994046f;
        out[0] = ((float)NPAIR * logd - Spair / T) / (LN10 * (float)(NROWS / 2));
    }
}

// ---------------------------------------------------------------------------
// Launcher. Workspace layout (~17 MB):
//   [0, 16MB)            : zn  (FP16, 16384 x 512)  -- L2-resident
//   [16MB, +64KB)        : per-block GEMM partials (128*128 floats)
//   [16MB+64KB, +64KB)   : per-row diag exp terms (16384 floats)
//   [16MB+128KB, +~16KB) : pair dots (4097 floats)
// ---------------------------------------------------------------------------
extern "C" void kernel_launch(void* const* d_in, const int* in_sizes, int n_in,
                              void* d_out, int out_size, void* d_ws, size_t ws_size,
                              hipStream_t stream) {
    (void)in_sizes; (void)n_in; (void)out_size; (void)ws_size;

    const float* y    = (const float*)d_in[0];
    const float* temp = (const float*)d_in[1];
    float* out = (float*)d_out;

    char* ws = (char*)d_ws;
    _Float16* zn = (_Float16*)ws;
    float* gp    = (float*)(ws + (size_t)16 * 1024 * 1024);
    float* diag  = (float*)(ws + (size_t)16 * 1024 * 1024 + 64 * 1024);
    float* pd    = (float*)(ws + (size_t)16 * 1024 * 1024 + 128 * 1024);

    normalize_rows<<<NROWS / 8, 256, 0, stream>>>(y, temp, zn, diag);

    dim3 grid(NROWS / BN, NROWS / BM);   // 128 x 128 blocks, lower triangle exits early
    gram_exp_sum<<<grid, 256, 0, stream>>>(zn, temp, gp);

    pair_dots<<<(NPAIR + 7) / 8, 256, 0, stream>>>(zn, pd);

    finalize<<<1, 256, 0, stream>>>(gp, (NROWS / BM) * (NROWS / BN), diag, pd, temp, out);
}
